// Attention_4398046511573
// MI455X (gfx1250) — compile-verified, matrix operand order checked by reading, not run on hardware
//
#include <hip/hip_runtime.h>
#include <hip/hip_bf16.h>
#include <stdint.h>

// ---------------------------------------------------------------------------
// Problem constants (match reference)
// ---------------------------------------------------------------------------
#define BB    512       // batch
#define TT    128       // time steps of encoder features
#define NIN   512       // input feature dim
#define HH    512       // hidden dim
#define NC    96        // num classes
#define STEPS 26        // decode steps
#define XDIM  1120      // NIN + NC + HH  (fused LSTM input)
#define G4H   2048      // 4*H

typedef __bf16 bf16;
typedef __attribute__((ext_vector_type(16))) __bf16 v16bf;
typedef __attribute__((ext_vector_type(8)))  float  v8f;

union FragBF { v16bf v; uint4 q[2]; };

__device__ __forceinline__ float sigmoidf_(float x) {
    return 1.0f / (1.0f + __expf(-x));
}

// ---------------------------------------------------------------------------
// Elementwise helper kernels
// ---------------------------------------------------------------------------
__global__ void cvt_f32_to_bf16(const float* __restrict__ src,
                                bf16* __restrict__ dst, long n) {
    long i = (long)blockIdx.x * blockDim.x + threadIdx.x;
    if (i < n) dst[i] = (bf16)src[i];
}

// Fused weight matrix: Wcat[r, 0:608] = W_ih[r, :], Wcat[r, 608:1120] = W_hh[r, :]
__global__ void build_wcat(const float* __restrict__ W_ih,   // [2048, 608]
                           const float* __restrict__ W_hh,   // [2048, 512]
                           bf16* __restrict__ Wcat) {        // [2048, 1120]
    long i = (long)blockIdx.x * blockDim.x + threadIdx.x;
    long n = (long)G4H * XDIM;
    if (i >= n) return;
    int r = (int)(i / XDIM);
    int c = (int)(i % XDIM);
    float v = (c < (NIN + NC)) ? W_ih[(long)r * (NIN + NC) + c]
                               : W_hh[(long)r * HH + (c - (NIN + NC))];
    Wcat[i] = (bf16)v;
}

__global__ void build_bcat(const float* __restrict__ b_ih,
                           const float* __restrict__ b_hh,
                           float* __restrict__ bcat) {
    int i = blockIdx.x * blockDim.x + threadIdx.x;
    if (i < G4H) bcat[i] = b_ih[i] + b_hh[i];
}

__global__ void zero_state(float* __restrict__ h, float* __restrict__ c,
                           bf16* __restrict__ hb, long n) {
    long i = (long)blockIdx.x * blockDim.x + threadIdx.x;
    if (i < n) { h[i] = 0.f; c[i] = 0.f; hb[i] = (bf16)0.f; }
}

// xcat[b, :] = [ context(512) | onehot(text[b,step]) (96) | h(512) ]  (bf16)
__global__ void build_xcat(const float* __restrict__ context, // [B, NIN] fp32
                           const int*   __restrict__ text,    // [B, STEPS]
                           const bf16*  __restrict__ hb,      // [B, H]
                           bf16* __restrict__ xcat,           // [B, XDIM]
                           int step) {
    long i = (long)blockIdx.x * blockDim.x + threadIdx.x;
    long n = (long)BB * XDIM;
    if (i >= n) return;
    int b = (int)(i / XDIM);
    int c = (int)(i % XDIM);
    float v;
    if (c < NIN) {
        v = context[(long)b * NIN + c];
    } else if (c < NIN + NC) {
        int cls = text[(long)b * STEPS + step];
        v = (c - NIN == cls) ? 1.0f : 0.0f;
    } else {
        v = (float)hb[(long)b * HH + (c - NIN - NC)];
    }
    xcat[i] = (bf16)v;
}

// LSTM pointwise cell (torch gate order i,f,g,o)
__global__ void lstm_cell(const float* __restrict__ gates, // [B, 4H]
                          float* __restrict__ h, float* __restrict__ c,
                          bf16* __restrict__ hb,           // [B, H]
                          bf16* __restrict__ hsb,          // [B, STEPS, H]
                          int step) {
    long idx = (long)blockIdx.x * blockDim.x + threadIdx.x;
    if (idx >= (long)BB * HH) return;
    int b = (int)(idx >> 9);
    int j = (int)(idx & (HH - 1));
    const float* g = gates + (long)b * G4H;
    float gi = g[j];
    float gf = g[HH + j];
    float gg = g[2 * HH + j];
    float go = g[3 * HH + j];
    float cn = sigmoidf_(gf) * c[idx] + sigmoidf_(gi) * tanhf(gg);
    float hn = sigmoidf_(go) * tanhf(cn);
    c[idx] = cn;
    h[idx] = hn;
    hb[idx] = (bf16)hn;
    hsb[((long)b * STEPS + step) * HH + j] = (bf16)hn;
}

// ---------------------------------------------------------------------------
// bf16 WMMA GEMM:  C[M,N] = A[M,K] * W[N,K]^T (+ bias[N])
// A, W row-major bf16.  One wave computes a 16 x (16*NT) strip: the A fragment
// is loaded once per K-step and reused across NT B-fragment WMMAs
// (2 + 2*NT b128 loads per NT WMMAs instead of 4 per WMMA).
// Fragment layouts per CDNA5 ISA 7.12.2 (16-bit A: lane<16 -> K {0..7,16..23},
// lane>=16 -> K {8..15,24..31}; B symmetric with N = lane&15).
// ---------------------------------------------------------------------------
template <int NT>
__global__ void gemm_bf16_wmma(const bf16* __restrict__ A,
                               const bf16* __restrict__ W,
                               const float* __restrict__ bias, // nullable
                               void* __restrict__ Cout,
                               int M, int N, int K,
                               int tilesNg, int outBf16) {
    long tid   = (long)blockIdx.x * blockDim.x + threadIdx.x;
    long wid   = tid >> 5;
    int  lane  = (int)(tid & 31);
    long nWave = (long)(M >> 4) * tilesNg;
    if (wid >= nWave) return;

    int tileM = (int)(wid / tilesNg) << 4;
    int tileN = (int)(wid % tilesNg) * (16 * NT);

    int lg  = lane >> 4;        // lane group 0/1
    int l15 = lane & 15;

    const bf16* aRow  = A + (long)(tileM + l15) * K + (lg << 3);
    const bf16* wBase = W + (long)(tileN + l15) * K + (lg << 3);
    const long  wStep = (long)16 * K;       // next 16-column group of C

    v8f acc[NT];
#pragma unroll
    for (int nt = 0; nt < NT; ++nt) acc[nt] = (v8f){};

    for (int k = 0; k < K; k += 32) {
        FragBF af;
        // elements 0..7 : K = k + 8*lg + {0..7}; elements 8..15: +16
        af.q[0] = *(const uint4*)(aRow + k);
        af.q[1] = *(const uint4*)(aRow + k + 16);
#pragma unroll
        for (int nt = 0; nt < NT; ++nt) {
            FragBF bf_;
            const bf16* wRow = wBase + (long)nt * wStep;
            bf_.q[0] = *(const uint4*)(wRow + k);
            bf_.q[1] = *(const uint4*)(wRow + k + 16);
            acc[nt] = __builtin_amdgcn_wmma_f32_16x16x32_bf16(
                false, af.v, false, bf_.v, (short)0, acc[nt], false, false);
        }
    }

#pragma unroll
    for (int nt = 0; nt < NT; ++nt) {
        int col = tileN + nt * 16 + l15;
        float badd = bias ? bias[col] : 0.0f;
#pragma unroll
        for (int r = 0; r < 8; ++r) {
            int row = tileM + r + (lg << 3);
            float v = acc[nt][r] + badd;
            if (outBf16) ((bf16*)Cout)[(long)row * N + col] = (bf16)v;
            else         ((float*)Cout)[(long)row * N + col] = v;
        }
    }
}

// ---------------------------------------------------------------------------
// Fused attention step: per block b:
//   emit[t] = sum_h tanh(feats[b,t,h] + h_proj[b,h]) * w_score[h]
//   alpha   = softmax_t(emit)
//   context[b,c] = sum_t alpha[t] * bh[b,t,c]
// ---------------------------------------------------------------------------
__global__ void attention_step(const bf16* __restrict__ feats,  // [B*T, H] bf16
                               const float* __restrict__ hproj, // [B, H]
                               const float* __restrict__ wscore,// [H]
                               const bf16* __restrict__ bh,     // [B*T, NIN] bf16
                               float* __restrict__ context) {   // [B, NIN]
    __shared__ float s_hp[HH];
    __shared__ float s_ws[HH];
    __shared__ float s_emit[TT];
    __shared__ float s_red[256];

    int b   = blockIdx.x;
    int tid = threadIdx.x;

    for (int i = tid; i < HH; i += 256) {
        s_hp[i] = hproj[(long)b * HH + i];
        s_ws[i] = wscore[i];
    }
    __syncthreads();

    // --- emit ---
    int wave = tid >> 5, lane = tid & 31;
    for (int t = wave; t < TT; t += 8) {
        const bf16* fr = feats + ((long)b * TT + t) * HH;
        float sum = 0.f;
        for (int h = lane; h < HH; h += 32)
            sum += tanhf((float)fr[h] + s_hp[h]) * s_ws[h];
#pragma unroll
        for (int off = 16; off > 0; off >>= 1)
            sum += __shfl_xor(sum, off, 32);
        if (lane == 0) s_emit[t] = sum;
    }
    __syncthreads();

    // --- softmax over T=128 ---
    float v = (tid < TT) ? s_emit[tid] : -3.402823e38f;
    s_red[tid] = v;
    __syncthreads();
    for (int off = 128; off > 0; off >>= 1) {
        if (tid < off) s_red[tid] = fmaxf(s_red[tid], s_red[tid + off]);
        __syncthreads();
    }
    float mx = s_red[0];
    __syncthreads();
    float e = (tid < TT) ? __expf(v - mx) : 0.f;
    s_red[tid] = e;
    __syncthreads();
    for (int off = 128; off > 0; off >>= 1) {
        if (tid < off) s_red[tid] += s_red[tid + off];
        __syncthreads();
    }
    float inv = 1.0f / s_red[0];
    if (tid < TT) s_emit[tid] = e * inv;
    __syncthreads();

    // --- context ---
    for (int c = tid; c < NIN; c += 256) {
        const bf16* bhb = bh + (long)b * TT * NIN + c;
        float acc = 0.f;
#pragma unroll 4
        for (int t = 0; t < TT; ++t)
            acc += s_emit[t] * (float)bhb[(long)t * NIN];
        context[(long)b * NIN + c] = acc;
    }
}

// ---------------------------------------------------------------------------
// Host-side launch helpers
// ---------------------------------------------------------------------------
static inline void launch_gemm(const bf16* A, const bf16* W, const float* bias,
                               void* C, int M, int N, int K, int outBf16,
                               hipStream_t s) {
    if ((N & 63) == 0) {
        const int NT = 4;
        int tilesNg = N / (16 * NT);
        long waves = (long)(M >> 4) * tilesNg;
        long threads = waves * 32;
        unsigned grid = (unsigned)((threads + 255) / 256);
        gemm_bf16_wmma<NT><<<grid, 256, 0, s>>>(A, W, bias, C, M, N, K, tilesNg, outBf16);
    } else if ((N % 96) == 0) {
        const int NT = 6;
        int tilesNg = N / (16 * NT);
        long waves = (long)(M >> 4) * tilesNg;
        long threads = waves * 32;
        unsigned grid = (unsigned)((threads + 255) / 256);
        gemm_bf16_wmma<NT><<<grid, 256, 0, s>>>(A, W, bias, C, M, N, K, tilesNg, outBf16);
    } else {
        const int NT = 1;
        int tilesNg = N / 16;
        long waves = (long)(M >> 4) * tilesNg;
        long threads = waves * 32;
        unsigned grid = (unsigned)((threads + 255) / 256);
        gemm_bf16_wmma<NT><<<grid, 256, 0, s>>>(A, W, bias, C, M, N, K, tilesNg, outBf16);
    }
}

static inline size_t align256(size_t x) { return (x + 255) & ~(size_t)255; }

extern "C" void kernel_launch(void* const* d_in, const int* in_sizes, int n_in,
                              void* d_out, int out_size, void* d_ws, size_t ws_size,
                              hipStream_t stream) {
    (void)in_sizes; (void)n_in; (void)out_size; (void)ws_size;

    const float* batch_hidden = (const float*)d_in[0];   // [B,T,NIN]
    const int*   text         = (const int*)  d_in[1];   // [B,STEPS]
    const float* W_i2h        = (const float*)d_in[2];   // [H,NIN]
    const float* W_h2h        = (const float*)d_in[3];   // [H,H]
    const float* b_h2h        = (const float*)d_in[4];   // [H]
    const float* w_score      = (const float*)d_in[5];   // [1,H]
    const float* W_ih         = (const float*)d_in[6];   // [4H, NIN+NC]
    const float* b_ih         = (const float*)d_in[7];   // [4H]
    const float* W_hh         = (const float*)d_in[8];   // [4H, H]
    const float* b_hh         = (const float*)d_in[9];   // [4H]
    const float* W_out        = (const float*)d_in[10];  // [NC, H]
    const float* b_out        = (const float*)d_in[11];  // [NC]
    float* out = (float*)d_out;                          // [B, STEPS, NC]

    // ---- workspace partition (bump allocator) ----
    char* ws = (char*)d_ws;
    size_t off = 0;
    auto alloc = [&](size_t bytes) -> char* {
        char* p = ws + off;
        off += align256(bytes);
        return p;
    };
    bf16*  bh_bf   = (bf16*) alloc((size_t)BB * TT * NIN * 2);   // 67 MB
    bf16*  feats   = (bf16*) alloc((size_t)BB * TT * HH  * 2);   // 67 MB
    bf16*  wi2h_b  = (bf16*) alloc((size_t)HH * NIN * 2);
    bf16*  wh2h_b  = (bf16*) alloc((size_t)HH * HH  * 2);
    bf16*  wcat_b  = (bf16*) alloc((size_t)G4H * XDIM * 2);
    float* bcat    = (float*)alloc((size_t)G4H * 4);
    bf16*  wout_b  = (bf16*) alloc((size_t)NC * HH * 2);
    float* h_st    = (float*)alloc((size_t)BB * HH * 4);
    float* c_st    = (float*)alloc((size_t)BB * HH * 4);
    bf16*  h_bf    = (bf16*) alloc((size_t)BB * HH * 2);
    float* h_proj  = (float*)alloc((size_t)BB * HH * 4);
    float* context = (float*)alloc((size_t)BB * NIN * 4);
    bf16*  xcat    = (bf16*) alloc((size_t)BB * XDIM * 2);
    float* gates   = (float*)alloc((size_t)BB * G4H * 4);
    bf16*  hs_bf   = (bf16*) alloc((size_t)BB * STEPS * HH * 2); // 13.6 MB

    // ---- pre-pass conversions ----
    {
        long n = (long)BB * TT * NIN;
        cvt_f32_to_bf16<<<(unsigned)((n + 255) / 256), 256, 0, stream>>>(batch_hidden, bh_bf, n);
        n = (long)HH * NIN;
        cvt_f32_to_bf16<<<(unsigned)((n + 255) / 256), 256, 0, stream>>>(W_i2h, wi2h_b, n);
        n = (long)HH * HH;
        cvt_f32_to_bf16<<<(unsigned)((n + 255) / 256), 256, 0, stream>>>(W_h2h, wh2h_b, n);
        n = (long)NC * HH;
        cvt_f32_to_bf16<<<(unsigned)((n + 255) / 256), 256, 0, stream>>>(W_out, wout_b, n);
        n = (long)G4H * XDIM;
        build_wcat<<<(unsigned)((n + 255) / 256), 256, 0, stream>>>(W_ih, W_hh, wcat_b);
        build_bcat<<<(G4H + 255) / 256, 256, 0, stream>>>(b_ih, b_hh, bcat);
        n = (long)BB * HH;
        zero_state<<<(unsigned)((n + 255) / 256), 256, 0, stream>>>(h_st, c_st, h_bf, n);
    }

    // feats_proj = batch_hidden @ W_i2h^T   -> bf16 [B*T, H]
    launch_gemm(bh_bf, wi2h_b, nullptr, feats, BB * TT, HH, NIN, /*bf16 out*/1, stream);

    // ---- sequential decode ----
    for (int step = 0; step < STEPS; ++step) {
        // h_proj = h @ W_h2h^T + b_h2h
        launch_gemm(h_bf, wh2h_b, b_h2h, h_proj, BB, HH, HH, 0, stream);

        // attention -> context
        attention_step<<<BB, 256, 0, stream>>>(feats, h_proj, w_score, bh_bf, context);

        // x = [context | onehot | h]
        {
            long n = (long)BB * XDIM;
            build_xcat<<<(unsigned)((n + 255) / 256), 256, 0, stream>>>(context, text, h_bf, xcat, step);
        }

        // gates = x @ [W_ih|W_hh]^T + (b_ih + b_hh)
        launch_gemm(xcat, wcat_b, bcat, gates, BB, G4H, XDIM, 0, stream);

        // LSTM cell
        {
            long n = (long)BB * HH;
            lstm_cell<<<(unsigned)((n + 255) / 256), 256, 0, stream>>>(gates, h_st, c_st, h_bf, hs_bf, step);
        }
    }

    // probs = hs @ W_out^T + b_out   -> fp32 [B*STEPS, NC]
    launch_gemm(hs_bf, wout_b, b_out, out, BB * STEPS, NC, HH, 0, stream);
}
